// ProteinMPNN_13941463843363
// MI455X (gfx1250) — matrix operand order repaired and checked by reading, not compiled, not run on hardware
//
#include <hip/hip_runtime.h>
#include <hip/hip_bf16.h>
#include <math.h>

// ============================================================================
// ProteinMPNN forward on MI455X (gfx1250): WMMA f16 GEMM pipeline.
//
// Shapes: B=4, N=1024, K=48, C=128, EF=128, VOCAB=21, 3 enc + 3 dec layers.
// Compute-bound (~320 GFLOP vs ~0.5GB traffic) -> v_wmma_f32_16x16x32_f16
// with LDS-staged per-node h_EV tiles, GELU fused into f16 tile stores,
// LayerNorm fused reductions, weights pre-packed into B-fragment layout.
//
// Input pytree order (jax dict flatten = sorted keys):
//   0:X 1:S 2:mask 3:E_idx 4:noise
//   5:W_e_b 6:W_e_w 7:W_out_b 8:W_out_w 9:W_s
//   10+l*14 (dec layer l): W1_b,W1_w,W2_b,W2_w,W3_b,W3_w,Wi_b,Wi_w,Wo_b,Wo_w,
//                          bn1,bn2,g1,g2
//   52+l*22 (enc layer l): W11_b,W11_w,W12_b,W12_w,W13_b,W13_w,W1_b,W1_w,
//                          W2_b,W2_w,W3_b,W3_w,Wi_b,Wi_w,Wo_b,Wo_w,
//                          bn1,bn2,bn3,g1,g2,g3
// ============================================================================

#define DEVI __device__ __forceinline__

typedef __attribute__((ext_vector_type(16))) _Float16 v16h;
typedef __attribute__((ext_vector_type(8)))  _Float16 v8h;
typedef __attribute__((ext_vector_type(8)))  float    v8f;

namespace {
constexpr int Bc = 4, Nn = 1024, Kn = 48, Cc = 128, VOC = 21;
constexpr int BN = Bc * Nn;
constexpr float INV_SCALE = 1.f / 30.f;
constexpr float EPS = 1e-5f;

DEVI int lane() { return (int)(threadIdx.x & 31u); }

DEVI float gelu(float x) { return 0.5f * x * (1.f + erff(x * 0.7071067811865475f)); }

// A fragment (16x32 f16 tile, row-major in LDS).  Documented 16-bit A layout:
// lane l (row = l&15), element j -> k = (j>>3)*16 + ((l>>4)<<3) + (j&7).
// Elements 0..7 and 8..15 are each K-contiguous -> two 16B loads per lane.
DEVI v16h fragA_lds(const _Float16* tile, int stride) {
  int l = lane();
  const _Float16* p = tile + (l & 15) * stride + ((l >> 4) << 3);
  v8h lo = *(const v8h*)(p);
  v8h hi = *(const v8h*)(p + 16);
  return __builtin_shufflevector(lo, hi, 0, 1, 2, 3, 4, 5, 6, 7,
                                 8, 9, 10, 11, 12, 13, 14, 15);
}

// B fragment: weights pre-packed so lane l's 16 halves are contiguous
// (32 bytes): one aligned v16h load per 32x16 tile.
DEVI v16h fragB_pk(const _Float16* tile) {
  return *(const v16h*)(tile + lane() * 16);
}

// One 16x16 D tile accumulated over ksteps K-chunks of 32.
DEVI v8f wmma_tile(const _Float16* Alds, int lda,
                   const _Float16* Bpk, int bstride, int ksteps, v8f acc) {
  for (int kt = 0; kt < ksteps; ++kt) {
    v16h a = fragA_lds(Alds + kt * 32, lda);
    v16h b = fragB_pk(Bpk + kt * bstride);
    acc = __builtin_amdgcn_wmma_f32_16x16x32_f16(
        false, a, false, b, (short)0, acc, false, false);
  }
  return acc;
}

// D layout: element v of lane l -> row m = v + ((l>>4)<<3), col n = l&15.
DEVI void store_gelu_f16(v8f d, float bias, _Float16* out, int stride) {
  int l = lane(), n = l & 15, mo = (l >> 4) << 3;
#pragma unroll
  for (int v = 0; v < 8; ++v)
    out[(mo + v) * stride + n] = (_Float16)gelu(d[v] + bias);
}

// Sum of `v` over threads 0..127 (all blockDim=256 threads must call).
DEVI float block_sum128(float v, float* buf) {
  int t = (int)threadIdx.x;
  if (t < 128) buf[t] = v;
  __syncthreads();
  if (t < 32) buf[t] = buf[t] + buf[t + 32] + buf[t + 64] + buf[t + 96];
  __syncthreads();
  if (t == 0) {
    float s = 0.f;
    for (int i = 0; i < 32; ++i) s += buf[i];
    buf[0] = s;
  }
  __syncthreads();
  float s = buf[0];
  __syncthreads();
  return s;
}
}  // namespace

// ---------------------------------------------------------------------------
// Pack f32 [K x N] weight into WMMA-B fragment tiles: [K/32][Npad/16][32][16],
// element (tile t, lane l, j) = W[kt*32 + ((j>>3)<<4) + ((l>>4)<<3) + (j&7)][nt*16 + (l&15)].
// ---------------------------------------------------------------------------
__global__ void k_pack(const float* __restrict__ src, _Float16* __restrict__ dst,
                       int K, int N, int Npad) {
  int total = (K / 32) * (Npad / 16) * 512;
  int nt16 = Npad / 16;
  for (int i = blockIdx.x * blockDim.x + threadIdx.x; i < total;
       i += gridDim.x * blockDim.x) {
    int t = i >> 9, r = i & 511, l = r >> 4, j = r & 15;
    int nt = t % nt16, kt = t / nt16;
    int n = nt * 16 + (l & 15);
    int k = kt * 32 + ((j >> 3) << 4) + ((l >> 4) << 3) + (j & 7);
    dst[i] = (_Float16)((n < N) ? src[k * N + n] : 0.f);
  }
}

// decoding_order = argsort((mask+1e-4)*|noise|) ; rank = argsort(order).
__global__ void k_rank(const float* __restrict__ mask,
                       const float* __restrict__ noise, int* __restrict__ rankv) {
  __shared__ float key[Nn];
  int b = blockIdx.x;
  for (int i = threadIdx.x; i < Nn; i += blockDim.x)
    key[i] = (mask[b * Nn + i] + 1e-4f) * fabsf(noise[b * Nn + i]);
  __syncthreads();
  for (int i = threadIdx.x; i < Nn; i += blockDim.x) {
    float ki = key[i];
    int r = 0;
    for (int j = 0; j < Nn; ++j) {
      float kj = key[j];
      r += (kj < ki) || (kj == ki && j < i);
    }
    rankv[b * Nn + i] = r;
  }
}

__global__ void k_init(float* __restrict__ hV, float* __restrict__ hS,
                       const int* __restrict__ S, const float* __restrict__ Ws) {
  int i = blockIdx.x * blockDim.x + threadIdx.x;
  if (i < BN * Cc) {
    hV[i] = 0.f;
    int node = i >> 7, c = i & 127;
    hS[i] = Ws[S[node] * Cc + c];
  }
}

// h_E = X @ W_e + b ; rows = B*N*K, 16 rows/block, 4 waves x 2 col-tiles.
__global__ void __launch_bounds__(128)
k_edge_embed(const float* __restrict__ X, const _Float16* __restrict__ Wpk,
             const float* __restrict__ bvec, float* __restrict__ hE) {
  __shared__ __attribute__((aligned(32))) _Float16 A[16 * 128];
  int r0 = blockIdx.x * 16;
  for (int i = threadIdx.x; i < 16 * 128; i += 128)
    A[i] = (_Float16)X[r0 * 128 + i];
  __syncthreads();
  int w = (int)(threadIdx.x >> 5);
  for (int s = 0; s < 2; ++s) {
    int nt = w * 2 + s;
    v8f acc = {};
    acc = wmma_tile(A, 128, Wpk + nt * 512, 8 * 512, 4, acc);
    int l = lane(), n = nt * 16 + (l & 15), mo = (l >> 4) << 3;
    float bia = bvec[n];
#pragma unroll
    for (int v = 0; v < 8; ++v) hE[(r0 + mo + v) * Cc + n] = acc[v] + bia;
  }
}

// Encoder node-message: h_EV[48x384] -> W1(gelu) -> W2(gelu) -> W3,
// * mask_attend, sum_k /30, residual, LN(g1,bn1) -> hVout.
__global__ void __launch_bounds__(256)
k_enc_msg(const float* __restrict__ hVin, const float* __restrict__ hE,
          const int* __restrict__ Eidx, const float* __restrict__ mask,
          const _Float16* __restrict__ W1, const float* __restrict__ b1,
          const _Float16* __restrict__ W2, const float* __restrict__ b2,
          const _Float16* __restrict__ W3, const float* __restrict__ b3,
          const float* __restrict__ g, const float* __restrict__ bn,
          float* __restrict__ hVout) {
  __shared__ __attribute__((aligned(32))) _Float16 hEV[Kn * 384];
  __shared__ __attribute__((aligned(32))) _Float16 h1[Kn * 128];
  __shared__ __attribute__((aligned(32))) _Float16 h2[Kn * 128];
  __shared__ float att[Kn];
  __shared__ int nb[Kn];
  __shared__ float vbuf[128];
  float* msg = (float*)hEV;  // 48*128 f32 = 24KB, aliases hEV (free post-GEMM1)

  const int node = blockIdx.x, bI = node >> 10;
  for (int e = threadIdx.x; e < Kn; e += blockDim.x) {
    int j = Eidx[node * Kn + e];
    nb[e] = j;
    att[e] = mask[node] * mask[bI * Nn + j];
  }
  __syncthreads();
  for (int i = threadIdx.x; i < Kn * 384; i += blockDim.x) {
    int e = i / 384, c = i - e * 384;
    float v;
    if (c < 128)       v = hVin[node * Cc + c];
    else if (c < 256)  v = hE[(node * Kn + e) * Cc + (c - 128)];
    else               v = hVin[(bI * Nn + nb[e]) * Cc + (c - 256)];
    hEV[i] = (_Float16)v;
  }
  __syncthreads();
  const int w = (int)(threadIdx.x >> 5);
  for (int mt = 0; mt < 3; ++mt) {
    v8f acc = {};
    acc = wmma_tile(hEV + mt * 16 * 384, 384, W1 + w * 512, 8 * 512, 12, acc);
    store_gelu_f16(acc, b1[w * 16 + (lane() & 15)], h1 + mt * 16 * 128 + w * 16, 128);
  }
  __syncthreads();
  for (int mt = 0; mt < 3; ++mt) {
    v8f acc = {};
    acc = wmma_tile(h1 + mt * 16 * 128, 128, W2 + w * 512, 8 * 512, 4, acc);
    store_gelu_f16(acc, b2[w * 16 + (lane() & 15)], h2 + mt * 16 * 128 + w * 16, 128);
  }
  __syncthreads();
  for (int mt = 0; mt < 3; ++mt) {
    v8f acc = {};
    acc = wmma_tile(h2 + mt * 16 * 128, 128, W3 + w * 512, 8 * 512, 4, acc);
    int l = lane(), n = w * 16 + (l & 15), mo = (l >> 4) << 3;
    float bia = b3[n];
#pragma unroll
    for (int v = 0; v < 8; ++v) {
      int e = mt * 16 + mo + v;
      msg[e * 128 + n] = (acc[v] + bia) * att[e];
    }
  }
  __syncthreads();
  int c = (int)threadIdx.x;
  float val = 0.f;
  if (c < 128) {
    float s = 0.f;
    for (int e = 0; e < Kn; ++e) s += msg[e * 128 + c];
    val = hVin[node * Cc + c] + s * INV_SCALE;
  }
  float mu = block_sum128(val, vbuf) * (1.f / 128.f);
  float d = (c < 128) ? (val - mu) : 0.f;
  float var = block_sum128(d * d, vbuf) * (1.f / 128.f);
  if (c < 128)
    hVout[node * Cc + c] = (val - mu) * rsqrtf(var + EPS) * g[c] + bn[c];
}

// Encoder edge update: h_E = LN(h_E + MLP3(h_EV; W11..W13), g3, bn3).
__global__ void __launch_bounds__(256)
k_enc_edge(const float* __restrict__ hV, float* __restrict__ hE,
           const int* __restrict__ Eidx,
           const _Float16* __restrict__ W1, const float* __restrict__ b1,
           const _Float16* __restrict__ W2, const float* __restrict__ b2,
           const _Float16* __restrict__ W3, const float* __restrict__ b3,
           const float* __restrict__ g, const float* __restrict__ bn) {
  __shared__ __attribute__((aligned(32))) _Float16 hEV[Kn * 384];
  __shared__ __attribute__((aligned(32))) _Float16 h1[Kn * 128];
  __shared__ int nb[Kn];
  __shared__ float scr[16][16];
  _Float16* h2 = hEV;                  // bytes [0,12288): free after GEMM1
  float* outv = (float*)(hEV + 6144);  // bytes [12288,36864)

  const int node = blockIdx.x, bI = node >> 10;
  for (int e = threadIdx.x; e < Kn; e += blockDim.x) nb[e] = Eidx[node * Kn + e];
  __syncthreads();
  for (int i = threadIdx.x; i < Kn * 384; i += blockDim.x) {
    int e = i / 384, c = i - e * 384;
    float v;
    if (c < 128)       v = hV[node * Cc + c];
    else if (c < 256)  v = hE[(node * Kn + e) * Cc + (c - 128)];
    else               v = hV[(bI * Nn + nb[e]) * Cc + (c - 256)];
    hEV[i] = (_Float16)v;
  }
  __syncthreads();
  const int w = (int)(threadIdx.x >> 5);
  for (int mt = 0; mt < 3; ++mt) {
    v8f acc = {};
    acc = wmma_tile(hEV + mt * 16 * 384, 384, W1 + w * 512, 8 * 512, 12, acc);
    store_gelu_f16(acc, b1[w * 16 + (lane() & 15)], h1 + mt * 16 * 128 + w * 16, 128);
  }
  __syncthreads();
  for (int mt = 0; mt < 3; ++mt) {
    v8f acc = {};
    acc = wmma_tile(h1 + mt * 16 * 128, 128, W2 + w * 512, 8 * 512, 4, acc);
    store_gelu_f16(acc, b2[w * 16 + (lane() & 15)], h2 + mt * 16 * 128 + w * 16, 128);
  }
  __syncthreads();
  for (int mt = 0; mt < 3; ++mt) {
    v8f acc = {};
    acc = wmma_tile(h2 + mt * 16 * 128, 128, W3 + w * 512, 8 * 512, 4, acc);
    int l = lane(), n = w * 16 + (l & 15), mo = (l >> 4) << 3;
    float bia = b3[n];
#pragma unroll
    for (int v = 0; v < 8; ++v) outv[(mt * 16 + mo + v) * 128 + n] = acc[v] + bia;
  }
  __syncthreads();
  // Per-edge-row LayerNorm; 16 rows per pass, 16 threads x 8 cols per row.
  const int sub = (int)(threadIdx.x & 15);
  for (int pass = 0; pass < 3; ++pass) {
    int rr = (int)(threadIdx.x >> 4);
    int r = pass * 16 + rr;
    float loc[8], s = 0.f;
#pragma unroll
    for (int q = 0; q < 8; ++q) {
      int cc = sub * 8 + q;
      float x = hE[(node * Kn + r) * Cc + cc] + outv[r * 128 + cc];
      loc[q] = x;
      s += x;
    }
    scr[rr][sub] = s;
    __syncthreads();
    float mu = 0.f;
    for (int i2 = 0; i2 < 16; ++i2) mu += scr[rr][i2];
    mu *= (1.f / 128.f);
    __syncthreads();
    s = 0.f;
#pragma unroll
    for (int q = 0; q < 8; ++q) { float dd = loc[q] - mu; s += dd * dd; }
    scr[rr][sub] = s;
    __syncthreads();
    float var = 0.f;
    for (int i2 = 0; i2 < 16; ++i2) var += scr[rr][i2];
    var *= (1.f / 128.f);
    float rs = rsqrtf(var + EPS);
#pragma unroll
    for (int q = 0; q < 8; ++q) {
      int cc = sub * 8 + q;
      hE[(node * Kn + r) * Cc + cc] = (loc[q] - mu) * rs * g[cc] + bn[cc];
    }
    __syncthreads();
  }
}

// Decoder node-message: h_EV[48x512] (autoregressive-masked concat) -> MLP3,
// sum_k /30, residual, LN(g1,bn1).
__global__ void __launch_bounds__(256)
k_dec_msg(const float* __restrict__ hVcur, const float* __restrict__ hVenc,
          const float* __restrict__ hE, const float* __restrict__ hS,
          const int* __restrict__ Eidx, const int* __restrict__ rankv,
          const float* __restrict__ mask,
          const _Float16* __restrict__ W1, const float* __restrict__ b1,
          const _Float16* __restrict__ W2, const float* __restrict__ b2,
          const _Float16* __restrict__ W3, const float* __restrict__ b3,
          const float* __restrict__ g, const float* __restrict__ bn,
          float* __restrict__ hVout) {
  __shared__ __attribute__((aligned(32))) _Float16 hEV[Kn * 512];
  __shared__ __attribute__((aligned(32))) _Float16 h1[Kn * 128];
  __shared__ float mb[Kn], fw[Kn];
  __shared__ int nb[Kn];
  __shared__ float vbuf[128];
  _Float16* h2 = hEV;                 // bytes [0,12288)
  float* msg = (float*)(hEV + 6144);  // bytes [12288,36864)

  const int node = blockIdx.x, bI = node >> 10;
  for (int e = threadIdx.x; e < Kn; e += blockDim.x) {
    int j = Eidx[node * Kn + e];
    nb[e] = j;
    float a = (rankv[node] > rankv[bI * Nn + j]) ? 1.f : 0.f;
    float m1 = mask[node];
    mb[e] = m1 * a;
    fw[e] = m1 * (1.f - a);
  }
  __syncthreads();
  for (int i = threadIdx.x; i < Kn * 512; i += blockDim.x) {
    int e = i >> 9, c = i & 511;
    float v;
    if (c < 128)       v = hVcur[node * Cc + c];
    else if (c < 256)  v = (mb[e] + fw[e]) * hE[(node * Kn + e) * Cc + (c - 128)];
    else if (c < 384)  v = mb[e] * hS[(bI * Nn + nb[e]) * Cc + (c - 256)];
    else               v = mb[e] * hVcur[(bI * Nn + nb[e]) * Cc + (c - 384)] +
                           fw[e] * hVenc[(bI * Nn + nb[e]) * Cc + (c - 384)];
    hEV[i] = (_Float16)v;
  }
  __syncthreads();
  const int w = (int)(threadIdx.x >> 5);
  for (int mt = 0; mt < 3; ++mt) {
    v8f acc = {};
    acc = wmma_tile(hEV + mt * 16 * 512, 512, W1 + w * 512, 8 * 512, 16, acc);
    store_gelu_f16(acc, b1[w * 16 + (lane() & 15)], h1 + mt * 16 * 128 + w * 16, 128);
  }
  __syncthreads();
  for (int mt = 0; mt < 3; ++mt) {
    v8f acc = {};
    acc = wmma_tile(h1 + mt * 16 * 128, 128, W2 + w * 512, 8 * 512, 4, acc);
    store_gelu_f16(acc, b2[w * 16 + (lane() & 15)], h2 + mt * 16 * 128 + w * 16, 128);
  }
  __syncthreads();
  for (int mt = 0; mt < 3; ++mt) {
    v8f acc = {};
    acc = wmma_tile(h2 + mt * 16 * 128, 128, W3 + w * 512, 8 * 512, 4, acc);
    int l = lane(), n = w * 16 + (l & 15), mo = (l >> 4) << 3;
    float bia = b3[n];
#pragma unroll
    for (int v = 0; v < 8; ++v) msg[(mt * 16 + mo + v) * 128 + n] = acc[v] + bia;
  }
  __syncthreads();
  int c = (int)threadIdx.x;
  float val = 0.f;
  if (c < 128) {
    float s = 0.f;
    for (int e = 0; e < Kn; ++e) s += msg[e * 128 + c];
    val = hVcur[node * Cc + c] + s * INV_SCALE;
  }
  float mu = block_sum128(val, vbuf) * (1.f / 128.f);
  float d = (c < 128) ? (val - mu) : 0.f;
  float var = block_sum128(d * d, vbuf) * (1.f / 128.f);
  if (c < 128)
    hVout[node * Cc + c] = (val - mu) * rsqrtf(var + EPS) * g[c] + bn[c];
}

// Position-wise FFN: hV = mask * LN(hV + Wo·gelu(Wi·hV + bi) + bo, g, bn).
// 16 node-rows per block; in-place safe (block owns its rows).
__global__ void __launch_bounds__(256)
k_ffn(const float* __restrict__ hVin, float* __restrict__ hVout,
      const float* __restrict__ mask,
      const _Float16* __restrict__ Wi, const float* __restrict__ bi,
      const _Float16* __restrict__ Wo, const float* __restrict__ bo,
      const float* __restrict__ g, const float* __restrict__ bn) {
  __shared__ __attribute__((aligned(32))) _Float16 A[16 * 128];
  __shared__ __attribute__((aligned(32))) _Float16 H[16 * 512];
  __shared__ float outv[16 * 128];
  __shared__ float scr[16][16];
  const int r0 = blockIdx.x * 16;
  for (int i = threadIdx.x; i < 16 * 128; i += blockDim.x)
    A[i] = (_Float16)hVin[r0 * Cc + i];
  __syncthreads();
  const int w = (int)(threadIdx.x >> 5);
  for (int s = 0; s < 4; ++s) {
    int nt = w * 4 + s;
    v8f acc = {};
    acc = wmma_tile(A, 128, Wi + nt * 512, 32 * 512, 4, acc);
    store_gelu_f16(acc, bi[nt * 16 + (lane() & 15)], H + nt * 16, 512);
  }
  __syncthreads();
  {
    v8f acc = {};
    acc = wmma_tile(H, 512, Wo + w * 512, 8 * 512, 16, acc);
    int l = lane(), n = w * 16 + (l & 15), mo = (l >> 4) << 3;
    float bia = bo[n];
#pragma unroll
    for (int v = 0; v < 8; ++v) outv[(mo + v) * 128 + n] = acc[v] + bia;
  }
  __syncthreads();
  const int r = (int)(threadIdx.x >> 4), sub = (int)(threadIdx.x & 15);
  float loc[8], s = 0.f;
#pragma unroll
  for (int q = 0; q < 8; ++q) {
    int cc = sub * 8 + q;
    float x = hVin[(r0 + r) * Cc + cc] + outv[r * 128 + cc];
    loc[q] = x;
    s += x;
  }
  scr[r][sub] = s;
  __syncthreads();
  float mu = 0.f;
  for (int i2 = 0; i2 < 16; ++i2) mu += scr[r][i2];
  mu *= (1.f / 128.f);
  __syncthreads();
  s = 0.f;
#pragma unroll
  for (int q = 0; q < 8; ++q) { float dd = loc[q] - mu; s += dd * dd; }
  scr[r][sub] = s;
  __syncthreads();
  float var = 0.f;
  for (int i2 = 0; i2 < 16; ++i2) var += scr[r][i2];
  var *= (1.f / 128.f);
  float rs = rsqrtf(var + EPS);
  float mk = mask[r0 + r];
#pragma unroll
  for (int q = 0; q < 8; ++q) {
    int cc = sub * 8 + q;
    hVout[(r0 + r) * Cc + cc] = mk * ((loc[q] - mu) * rs * g[cc] + bn[cc]);
  }
}

// Output head: logits = hV @ W_out (padded to 32 cols) + b ; softmax(21).
__global__ void __launch_bounds__(64)
k_out(const float* __restrict__ hV, const _Float16* __restrict__ Wpk,
      const float* __restrict__ bo, float* __restrict__ out) {
  __shared__ __attribute__((aligned(32))) _Float16 A[16 * 128];
  __shared__ float L[16 * 32];
  const int r0 = blockIdx.x * 16;
  for (int i = threadIdx.x; i < 16 * 128; i += blockDim.x)
    A[i] = (_Float16)hV[r0 * Cc + i];
  __syncthreads();
  const int w = (int)(threadIdx.x >> 5);
  v8f acc = {};
  acc = wmma_tile(A, 128, Wpk + w * 512, 2 * 512, 4, acc);
  int l = lane(), n = w * 16 + (l & 15), mo = (l >> 4) << 3;
  float bia = (n < VOC) ? bo[n] : 0.f;
#pragma unroll
  for (int v = 0; v < 8; ++v) L[(mo + v) * 32 + n] = acc[v] + bia;
  __syncthreads();
  if (threadIdx.x < 16) {
    int r = (int)threadIdx.x;
    float mx = -1e30f;
    for (int c2 = 0; c2 < VOC; ++c2) mx = fmaxf(mx, L[r * 32 + c2]);
    float ex[VOC], s = 0.f;
    for (int c2 = 0; c2 < VOC; ++c2) { ex[c2] = expf(L[r * 32 + c2] - mx); s += ex[c2]; }
    float inv = 1.f / s;
    for (int c2 = 0; c2 < VOC; ++c2) out[(r0 + r) * VOC + c2] = ex[c2] * inv;
  }
}

// ============================================================================
extern "C" void kernel_launch(void* const* d_in, const int* in_sizes, int n_in,
                              void* d_out, int out_size, void* d_ws, size_t ws_size,
                              hipStream_t stream) {
  (void)in_sizes; (void)n_in; (void)out_size; (void)ws_size;

  const float* X      = (const float*)d_in[0];
  const int*   S      = (const int*)  d_in[1];
  const float* mask   = (const float*)d_in[2];
  const int*   Eidx   = (const int*)  d_in[3];
  const float* noise  = (const float*)d_in[4];
  const float* We_b   = (const float*)d_in[5];
  const float* We_w   = (const float*)d_in[6];
  const float* Wout_b = (const float*)d_in[7];
  const float* Wout_w = (const float*)d_in[8];
  const float* Ws     = (const float*)d_in[9];
  auto decP = [&](int l, int k) { return (const float*)d_in[10 + l * 14 + k]; };
  auto encP = [&](int l, int k) { return (const float*)d_in[52 + l * 22 + k]; };

  char* base = (char*)d_ws;
  size_t off = 0;
  auto alloc = [&](size_t bytes) -> void* {
    void* p = base + off;
    off = (off + bytes + 255) & ~(size_t)255;
    return p;
  };

  float* hVa   = (float*)alloc((size_t)BN * Cc * 4);
  float* hVb   = (float*)alloc((size_t)BN * Cc * 4);
  float* hVenc = (float*)alloc((size_t)BN * Cc * 4);
  float* hS    = (float*)alloc((size_t)BN * Cc * 4);
  float* hE    = (float*)alloc((size_t)BN * Kn * Cc * 4);
  int*   rnk   = (int*)alloc((size_t)BN * 4);

  auto allocH = [&](size_t halves) { return (_Float16*)alloc(halves * 2); };
  _Float16* We_pk   = allocH(16384);
  _Float16* Wout_pk = allocH(4096);
  _Float16 *eW1[3], *eW2[3], *eW3[3], *eW11[3], *eW12[3], *eW13[3], *eWi[3], *eWo[3];
  _Float16 *dW1[3], *dW2[3], *dW3[3], *dWi[3], *dWo[3];
  for (int l = 0; l < 3; ++l) {
    eW1[l] = allocH(49152);  eW2[l] = allocH(16384);  eW3[l] = allocH(16384);
    eW11[l] = allocH(49152); eW12[l] = allocH(16384); eW13[l] = allocH(16384);
    eWi[l] = allocH(65536);  eWo[l] = allocH(65536);
  }
  for (int l = 0; l < 3; ++l) {
    dW1[l] = allocH(65536); dW2[l] = allocH(16384); dW3[l] = allocH(16384);
    dWi[l] = allocH(65536); dWo[l] = allocH(65536);
  }

  auto pack = [&](const float* src, _Float16* dst, int K, int N, int Npad) {
    int total = (K / 32) * (Npad / 16) * 512;
    k_pack<<<(total + 255) / 256, 256, 0, stream>>>(src, dst, K, N, Npad);
  };
  pack(We_w, We_pk, 128, 128, 128);
  pack(Wout_w, Wout_pk, 128, VOC, 32);
  for (int l = 0; l < 3; ++l) {
    pack(encP(l, 7),  eW1[l], 384, 128, 128);
    pack(encP(l, 9),  eW2[l], 128, 128, 128);
    pack(encP(l, 11), eW3[l], 128, 128, 128);
    pack(encP(l, 1),  eW11[l], 384, 128, 128);
    pack(encP(l, 3),  eW12[l], 128, 128, 128);
    pack(encP(l, 5),  eW13[l], 128, 128, 128);
    pack(encP(l, 13), eWi[l], 128, 512, 512);
    pack(encP(l, 15), eWo[l], 512, 128, 128);
    pack(decP(l, 1),  dW1[l], 512, 128, 128);
    pack(decP(l, 3),  dW2[l], 128, 128, 128);
    pack(decP(l, 5),  dW3[l], 128, 128, 128);
    pack(decP(l, 7),  dWi[l], 128, 512, 512);
    pack(decP(l, 9),  dWo[l], 512, 128, 128);
  }

  k_init<<<(BN * Cc + 255) / 256, 256, 0, stream>>>(hVa, hS, S, Ws);
  k_rank<<<Bc, 256, 0, stream>>>(mask, noise, rnk);
  k_edge_embed<<<(BN * Kn) / 16, 128, 0, stream>>>(X, We_pk, We_b, hE);

  float* cur = hVa;
  float* nxt = hVb;
  for (int l = 0; l < 3; ++l) {
    k_enc_msg<<<BN, 256, 0, stream>>>(cur, hE, Eidx, mask,
        eW1[l], encP(l, 6), eW2[l], encP(l, 8), eW3[l], encP(l, 10),
        encP(l, 19), encP(l, 16), nxt);
    k_ffn<<<BN / 16, 256, 0, stream>>>(nxt, nxt, mask,
        eWi[l], encP(l, 12), eWo[l], encP(l, 14), encP(l, 20), encP(l, 17));
    k_enc_edge<<<BN, 256, 0, stream>>>(nxt, hE, Eidx,
        eW11[l], encP(l, 0), eW12[l], encP(l, 2), eW13[l], encP(l, 4),
        encP(l, 21), encP(l, 18));
    float* t = cur; cur = nxt; nxt = t;
  }

  hipMemcpyAsync(hVenc, cur, (size_t)BN * Cc * 4, hipMemcpyDeviceToDevice, stream);

  for (int l = 0; l < 3; ++l) {
    k_dec_msg<<<BN, 256, 0, stream>>>(cur, hVenc, hE, hS, Eidx, rnk, mask,
        dW1[l], decP(l, 0), dW2[l], decP(l, 2), dW3[l], decP(l, 4),
        decP(l, 12), decP(l, 10), nxt);
    k_ffn<<<BN / 16, 256, 0, stream>>>(nxt, nxt, mask,
        dWi[l], decP(l, 6), dWo[l], decP(l, 8), decP(l, 13), decP(l, 11));
    float* t = cur; cur = nxt; nxt = t;
  }

  k_out<<<BN / 16, 64, 0, stream>>>(cur, Wout_pk, Wout_b, (float*)d_out);
}
